// HyperbolicWorldlines_45234595561960
// MI455X (gfx1250) — matrix operand order repaired and checked by reading, not compiled, not run on hardware
//
#include <hip/hip_runtime.h>
#include <math.h>

#define BB 16
#define NN 2048
#define DD 64
#define HH 128
#define NOBJ 8
#define NLVL 3
#define NSLOT 24
#define TAUV 0.1f
#define EPSC 1e-6f
#define BEPSC 1e-5f

typedef _Float16 f16;
typedef __attribute__((ext_vector_type(16))) _Float16 v16h;
typedef __attribute__((ext_vector_type(8)))  float    v8f;

// ---------------------------------------------------------------------------
// WMMA helpers (CDNA5 v_wmma_f32_16x16x32_f16)
// A layout (16-bit A 16x32): lane L: M = L&15, half = L>>4,
//   element e: K = ((e<8) ? e : e+8) + 8*half   (two contiguous 8-elem runs)
// B layout (16-bit B 32x16): lane L: N = L&15, half = L>>4,
//   element e: K = ((e<8) ? e : e+8) + 8*half   (two contiguous 8-elem runs)
// C/D f32 16x16: lane L: N = L&15; VGPR v: M = v + 8*(L>>4)
// ---------------------------------------------------------------------------
__device__ __forceinline__ v8f wmma16(v16h a, v16h b, v8f c) {
  return __builtin_amdgcn_wmma_f32_16x16x32_f16(false, a, false, b,
                                                (short)0, c, false, false);
}

__device__ __forceinline__ v16h loadA16(const f16* row, int half, int k0) {
  v16h a;
  const f16* p0 = row + k0 + 8 * half;
  const f16* p1 = row + k0 + 16 + 8 * half;
#pragma unroll
  for (int e = 0; e < 8; ++e) a[e] = p0[e];
#pragma unroll
  for (int e = 0; e < 8; ++e) a[8 + e] = p1[e];
  return a;
}

__device__ __forceinline__ v16h loadB16(const f16* colT, int half, int k0) {
  v16h b;
  const f16* p0 = colT + k0 + 8 * half;
  const f16* p1 = colT + k0 + 16 + 8 * half;
#pragma unroll
  for (int e = 0; e < 8; ++e) b[e] = p0[e];
#pragma unroll
  for (int e = 0; e < 8; ++e) b[8 + e] = p1[e];
  return b;
}

// sorted-ascending top-6 insert (registers only)
__device__ __forceinline__ void top6_insert(float (&top)[6], float vd) {
  if (vd < top[5]) {
    top[5] = vd;
#pragma unroll
    for (int q = 5; q > 0; --q) {
      if (top[q] < top[q - 1]) {
        float t = top[q]; top[q] = top[q - 1]; top[q - 1] = t;
      }
    }
  }
}

// ---------------------------------------------------------------------------
// Prep: x -> f16 + per-point squared norm
// ---------------------------------------------------------------------------
__global__ void k_prep_x(const float* __restrict__ x, f16* __restrict__ xh,
                         float* __restrict__ x2) {
  int i = blockIdx.x * blockDim.x + threadIdx.x;
  if (i >= BB * NN) return;
  const float* src = x + (size_t)i * DD;
  f16* dst = xh + (size_t)i * DD;
  float s = 0.f;
#pragma unroll
  for (int k = 0; k < DD; ++k) {
    float v = src[k];
    s += v * v;
    dst[k] = (f16)v;
  }
  x2[i] = s;
}

// transpose [fin,fout] f32 -> [fout,fin] f16 so B-matrix loads are contiguous
__global__ void k_transpose_f16(const float* __restrict__ w, f16* __restrict__ wt,
                                int fin, int fout) {
  int i = blockIdx.x * blockDim.x + threadIdx.x;
  if (i >= fin * fout) return;
  int n = i / fin, k = i % fin;
  wt[(size_t)n * fin + k] = (f16)w[(size_t)k * fout + n];
}

// ---------------------------------------------------------------------------
// Initial slots (unit obj dirs at 3 radii) + precomputed gh = old_h @ gru_wh^T + b
// (old_h == obj_dirs is constant across iterations in the reference)
// ---------------------------------------------------------------------------
__global__ void k_init(const float* __restrict__ obj_dirs,
                       const float* __restrict__ gru_wh,
                       const float* __restrict__ gru_bh,
                       float* __restrict__ slots, float* __restrict__ gh_pre) {
  __shared__ float dn[NOBJ];
  int tid = threadIdx.x;
  if (tid < NOBJ) {
    float s = 0.f;
    for (int h = 0; h < HH; ++h) {
      float v = obj_dirs[tid * HH + h];
      s += v * v;
    }
    dn[tid] = 1.0f / fmaxf(sqrtf(s), 1e-12f);
  }
  __syncthreads();
  const float radii[3] = {0.2f, 0.5f, 0.8f};
  for (int idx = tid; idx < NSLOT * HH; idx += blockDim.x) {
    int k = idx >> 7, h = idx & 127;
    int o = k / NLVL, l = k % NLVL;
    float v = obj_dirs[o * HH + h] * dn[o] * radii[l];
    for (int b = 0; b < BB; ++b)
      slots[((size_t)b * NSLOT + k) * HH + h] = v;
  }
  for (int idx = tid; idx < NOBJ * 3 * HH; idx += blockDim.x) {
    int o = idx / (3 * HH), j = idx % (3 * HH);
    float acc = gru_bh[j];
    const float* wr = gru_wh + (size_t)j * HH;
    const float* od = obj_dirs + o * HH;
    for (int h = 0; h < HH; ++h) acc += od[h] * wr[h];
    gh_pre[idx] = acc;
  }
}

// ---------------------------------------------------------------------------
// KNN density: WMMA x.x^T tiles, top-6 selection on SQUARED distances
// (sqrt is monotone; only 5 sqrts per row at the end).
// Single-wave workgroups: barriers degrade to S_NOP + ds-waits.
// Row stride 20 floats keeps each 8-value run 16B-aligned (ds_load_b128-able)
// and bank-conflict-free (20*r mod 64 distinct for r=0..15).
// grid = BB * (NN/16), block = 32. Two lanes cooperate per row (8 cols each).
// ---------------------------------------------------------------------------
__global__ void k_density(const f16* __restrict__ xh, const float* __restrict__ x2,
                          float* __restrict__ dens_out) {
  __shared__ __align__(16) float tile[16][20];
  __shared__ __align__(16) float mtop[32][6];
  int lane = threadIdx.x;
  int m = lane & 15, half = lane >> 4;
  int b = blockIdx.x / (NN / 16);
  int r0 = (blockIdx.x % (NN / 16)) * 16;
  const f16* xb = xh + (size_t)b * NN * DD;
  const float* x2b = x2 + (size_t)b * NN;

  const f16* arow = xb + (size_t)(r0 + m) * DD;
  v16h a0 = loadA16(arow, half, 0);
  v16h a1 = loadA16(arow, half, 32);
  float rx2[8];
#pragma unroll
  for (int v = 0; v < 8; ++v) rx2[v] = x2b[r0 + v + 8 * half];

  float top[6];
#pragma unroll
  for (int q = 0; q < 6; ++q) top[q] = 3.0e38f;

  for (int j0 = 0; j0 < NN; j0 += 16) {
    const f16* bcol = xb + (size_t)(j0 + m) * DD;
    __builtin_prefetch(bcol + 16 * DD, 0, 0);
    v16h b0 = loadB16(bcol, half, 0);
    v16h b1 = loadB16(bcol, half, 32);
    v8f c = {};
    c = wmma16(a0, b0, c);
    c = wmma16(a1, b1, c);
    float colx2 = x2b[j0 + m];
#pragma unroll
    for (int v = 0; v < 8; ++v) {
      int mr = v + 8 * half;
      tile[mr][m] = fmaxf(rx2[v] + colx2 - 2.0f * c[v], 0.f);  // squared dist
    }
    __syncthreads();
    // batch the 8 scan values into registers (one dscnt wait), then VALU scan
    float vals[8];
#pragma unroll
    for (int e = 0; e < 8; ++e) vals[e] = tile[m][8 * half + e];
#pragma unroll
    for (int e = 0; e < 8; ++e) top6_insert(top, vals[e]);
    __syncthreads();
  }
  // merge the two half-row top-6 lists
#pragma unroll
  for (int q = 0; q < 6; ++q) mtop[lane][q] = top[q];
  __syncthreads();
  if (lane < 16) {
    float ov[6];
#pragma unroll
    for (int q = 0; q < 6; ++q) ov[q] = mtop[lane + 16][q];
#pragma unroll
    for (int q = 0; q < 6; ++q) top6_insert(top, ov[q]);
    // top[0] is the self distance (~0); mean of sqrt of the next 5
    float md = (sqrtf(top[1]) + sqrtf(top[2]) + sqrtf(top[3]) +
                sqrtf(top[4]) + sqrtf(top[5])) * 0.2f;
    dens_out[(size_t)b * NN + r0 + lane] = tanhf(md);
  }
}

// ---------------------------------------------------------------------------
// Encoder: x -> LN/GELU -> LN -> tanh(hp) -> radius-normalized feats (f16 x2)
// grid = BB * (NN/16), block = 256 (8 waves each owning a 16-col slab)
// ---------------------------------------------------------------------------
__global__ void k_encoder(const f16* __restrict__ xh,
                          const float* __restrict__ fe_b1, const float* __restrict__ ln1_g,
                          const float* __restrict__ ln1_b,
                          const float* __restrict__ fe_b2, const float* __restrict__ ln2_g,
                          const float* __restrict__ ln2_b,
                          const float* __restrict__ hp_b,
                          const f16* __restrict__ w1t, const f16* __restrict__ w2t,
                          const f16* __restrict__ hpt,
                          const float* __restrict__ dens,
                          f16* __restrict__ feats_h, f16* __restrict__ featsT,
                          float* __restrict__ f2out, float* __restrict__ frout) {
  __shared__ float tbuf[16][HH + 1];
  __shared__ f16 hbuf[16][HH];
  __shared__ float rmu[16], rsg[16], rscale[16];
  int tid = threadIdx.x;
  int wave = tid >> 5, lane = tid & 31, m = lane & 15, half = lane >> 4;
  int b = blockIdx.x >> 7;
  int r0 = (blockIdx.x & 127) * 16;
  int col = wave * 16 + m;

  // stage A: t1 = x @ fe_w1 + b1   (K=64)
  const f16* arow = xh + ((size_t)b * NN + r0 + m) * DD;
  v16h a0 = loadA16(arow, half, 0);
  v16h a1 = loadA16(arow, half, 32);
  const f16* bt1 = w1t + (size_t)col * DD;
  v8f c = {};
  c = wmma16(a0, loadB16(bt1, half, 0), c);
  c = wmma16(a1, loadB16(bt1, half, 32), c);
  float bias1 = fe_b1[col];
#pragma unroll
  for (int v = 0; v < 8; ++v) tbuf[v + 8 * half][col] = c[v] + bias1;
  __syncthreads();
  if (tid < 16) {
    float s = 0, s2 = 0;
    for (int h = 0; h < HH; ++h) { float t = tbuf[tid][h]; s += t; s2 += t * t; }
    float mu = s * (1.0f / HH);
    rmu[tid] = mu;
    rsg[tid] = rsqrtf(fmaxf(s2 * (1.0f / HH) - mu * mu, 0.f) + 1e-5f);
  }
  __syncthreads();
  for (int idx = tid; idx < 16 * HH; idx += 256) {
    int r = idx >> 7, h = idx & 127;
    float t = (tbuf[r][h] - rmu[r]) * rsg[r] * ln1_g[h] + ln1_b[h];
    t = 0.5f * t * (1.0f + erff(t * 0.70710678118f));   // exact GELU
    hbuf[r][h] = (f16)t;
  }
  __syncthreads();

  // stage B: t2 = h @ fe_w2 + b2   (K=128)
  const f16* bt2 = w2t + (size_t)col * HH;
  v8f c2 = {};
#pragma unroll
  for (int ks = 0; ks < 4; ++ks)
    c2 = wmma16(loadA16(&hbuf[m][0], half, ks * 32), loadB16(bt2, half, ks * 32), c2);
  float bias2 = fe_b2[col];
#pragma unroll
  for (int v = 0; v < 8; ++v) tbuf[v + 8 * half][col] = c2[v] + bias2;
  __syncthreads();
  if (tid < 16) {
    float s = 0, s2 = 0;
    for (int h = 0; h < HH; ++h) { float t = tbuf[tid][h]; s += t; s2 += t * t; }
    float mu = s * (1.0f / HH);
    rmu[tid] = mu;
    rsg[tid] = rsqrtf(fmaxf(s2 * (1.0f / HH) - mu * mu, 0.f) + 1e-5f);
  }
  __syncthreads();
  for (int idx = tid; idx < 16 * HH; idx += 256) {
    int r = idx >> 7, h = idx & 127;
    float t = (tbuf[r][h] - rmu[r]) * rsg[r] * ln2_g[h] + ln2_b[h];
    hbuf[r][h] = (f16)t;                                 // enc (f16)
  }
  __syncthreads();

  // stage C: hyp = tanh(enc @ hp_w + hp_b)   (K=128)
  const f16* bt3 = hpt + (size_t)col * HH;
  v8f c3 = {};
#pragma unroll
  for (int ks = 0; ks < 4; ++ks)
    c3 = wmma16(loadA16(&hbuf[m][0], half, ks * 32), loadB16(bt3, half, ks * 32), c3);
  float bias3 = hp_b[col];
#pragma unroll
  for (int v = 0; v < 8; ++v)
    tbuf[v + 8 * half][col] = tanhf(c3[v] + bias3);
  __syncthreads();
  if (tid < 16) {
    float s2 = 0;
    for (int h = 0; h < HH; ++h) { float t = tbuf[tid][h]; s2 += t * t; }
    float fn = fmaxf(sqrtf(s2), 1e-6f);
    float d = dens[(size_t)b * NN + r0 + tid];
    float tr = (0.3f + 0.5f * (1.0f - d)) * 0.95f;       // final feat radius
    rscale[tid] = tr / fn;
    f2out[(size_t)b * NN + r0 + tid] = tr * tr;
    frout[(size_t)b * NN + r0 + tid] = tr;
  }
  __syncthreads();
  for (int idx = tid; idx < 16 * HH; idx += 256) {
    int r = idx >> 7, h = idx & 127;
    float v = tbuf[r][h] * rscale[r];
    feats_h[((size_t)b * NN + r0 + r) * HH + h] = (f16)v;
    featsT[((size_t)b * HH + h) * NN + r0 + r] = (f16)v;
  }
}

// ---------------------------------------------------------------------------
// zero helper
// ---------------------------------------------------------------------------
__global__ void k_zero(float* __restrict__ p, int n) {
  int i = blockIdx.x * blockDim.x + threadIdx.x;
  if (i < n) p[i] = 0.f;
}

// ---------------------------------------------------------------------------
// dots = slots @ feats^T via WMMA (rows padded 24 -> 32) + slot norms
// grid = BB * (NN/128), block = 256 (8 waves, one 16-col n-tile each)
// ---------------------------------------------------------------------------
__global__ void k_dots(const float* __restrict__ slots, const f16* __restrict__ feats_h,
                       float* __restrict__ dots, float* __restrict__ snorm) {
  __shared__ f16 sh[32][HH];
  int tid = threadIdx.x, wave = tid >> 5, lane = tid & 31;
  int m = lane & 15, half = lane >> 4;
  int b = blockIdx.x / (NN / 128);
  int n0 = (blockIdx.x % (NN / 128)) * 128 + wave * 16;
  const float* sb = slots + (size_t)b * NSLOT * HH;

  for (int idx = tid; idx < 32 * HH; idx += 256) {
    int k = idx >> 7, h = idx & 127;
    sh[k][h] = (k < NSLOT) ? (f16)sb[(size_t)k * HH + h] : (f16)0.f;
  }
  __syncthreads();
  if (tid < NSLOT) {
    float s = 0;
    for (int h = 0; h < HH; ++h) { float v = sb[(size_t)tid * HH + h]; s += v * v; }
    snorm[((size_t)b * NSLOT + tid) * 2 + 0] = s;
    snorm[((size_t)b * NSLOT + tid) * 2 + 1] = sqrtf(s);
  }

  const f16* bp = feats_h + ((size_t)b * NN + n0 + m) * HH;
  __builtin_prefetch(bp, 0, 0);
  v8f c0 = {}, c1 = {};
#pragma unroll
  for (int ks = 0; ks < 4; ++ks) {
    v16h bb = loadB16(bp, half, ks * 32);
    c0 = wmma16(loadA16(&sh[m][0], half, ks * 32), bb, c0);
    c1 = wmma16(loadA16(&sh[16 + m][0], half, ks * 32), bb, c1);
  }
  int n = n0 + m;
#pragma unroll
  for (int v = 0; v < 8; ++v) {
    int k0r = v + 8 * half;
    dots[((size_t)b * NSLOT + k0r) * NN + n] = c0[v];
    int k1r = 16 + v + 8 * half;
    if (k1r < NSLOT)
      dots[((size_t)b * NSLOT + k1r) * NN + n] = c1[v];
  }
}

// ---------------------------------------------------------------------------
// attn: hyperbolic logits + softmax over slots + attn row sums
// grid = BB * (NN/256), block = 256 (one point per thread)
// ---------------------------------------------------------------------------
__global__ void k_attn(const float* __restrict__ dots, const float* __restrict__ snorm,
                       const float* __restrict__ f2, const float* __restrict__ fr,
                       const float* __restrict__ dens, const float* __restrict__ rscale_p,
                       float* __restrict__ attn_out, float* __restrict__ attn_sum) {
  __shared__ float s2s[NSLOT], srs[NSLOT], lsum[NSLOT];
  int tid = threadIdx.x;
  int b = blockIdx.x >> 3;
  int n = ((blockIdx.x & 7) << 8) + tid;
  if (tid < NSLOT) {
    s2s[tid] = snorm[((size_t)b * NSLOT + tid) * 2 + 0];
    srs[tid] = snorm[((size_t)b * NSLOT + tid) * 2 + 1];
    lsum[tid] = 0.f;
  }
  __syncthreads();

  float f2v = f2[(size_t)b * NN + n];
  float frv = fr[(size_t)b * NN + n];
  float mod = rscale_p[0] * (dens[(size_t)b * NN + n] - 0.5f);
  float xn = fminf(f2v, 1.0f - BEPSC);
  const float base_sc[3] = {2.0f, 1.0f, 0.5f};
  float lg[NSLOT];
  float mx = -3e38f;
#pragma unroll
  for (int k = 0; k < NSLOT; ++k) {
    float dv = dots[((size_t)b * NSLOT + k) * NN + n];
    float s2v = s2s[k];
    float diff2 = fmaxf(f2v + s2v - 2.0f * dv, 0.f);
    float yn = fminf(s2v, 1.0f - BEPSC);
    float den = fmaxf((1.0f - xn) * (1.0f - yn), EPSC);
    float arg = fmaxf(1.0f + 2.0f * diff2 / den, 1.0f + EPSC);
    float hd = acoshf(arg);
    float rdiff = fabsf(frv - srs[k]);
    float sc = fminf(fmaxf(base_sc[k % 3] + mod, 0.3f), 3.0f);
    float l = (-hd - 3.0f * rdiff) * sc * (1.0f / TAUV);
    lg[k] = l;
    mx = fmaxf(mx, l);
  }
  float se = 0.f;
#pragma unroll
  for (int k = 0; k < NSLOT; ++k) { lg[k] = expf(lg[k] - mx); se += lg[k]; }
  float inv = 1.0f / se;
#pragma unroll
  for (int k = 0; k < NSLOT; ++k) {
    float a = lg[k] * inv;
    attn_out[((size_t)b * NSLOT + k) * NN + n] = a;
    atomicAdd(&lsum[k], a);
  }
  __syncthreads();
  if (tid < NSLOT) atomicAdd(&attn_sum[b * NSLOT + tid], lsum[tid]);
}

// ---------------------------------------------------------------------------
// upd = (attn / sum) @ feats via WMMA; slot rows padded 24 -> 32
// grid = BB, block = 256 (8 waves, one 16-col slab each, 2 row tiles)
// ---------------------------------------------------------------------------
__global__ void k_upd(const float* __restrict__ attn, const float* __restrict__ attn_sum,
                      const f16* __restrict__ featsT, float* __restrict__ upd) {
  __shared__ float inv[32];
  int tid = threadIdx.x, wave = tid >> 5, lane = tid & 31;
  int m = lane & 15, half = lane >> 4;
  int b = blockIdx.x;
  if (tid < 32)
    inv[tid] = (tid < NSLOT) ? 1.0f / (attn_sum[b * NSLOT + tid] + 1e-8f) : 0.f;
  __syncthreads();

  int hh = wave * 16 + m;
  const f16* btRow = featsT + ((size_t)b * HH + hh) * NN;
  const float* attb = attn + (size_t)b * NSLOT * NN;
  float inv0 = inv[m];
  float inv1 = inv[16 + m];
  const float* ar0 = attb + (size_t)m * NN;
  const float* ar1 = (16 + m < NSLOT) ? attb + (size_t)(16 + m) * NN : attb;

  v8f c0 = {}, c1 = {};
  for (int n0 = 0; n0 < NN; n0 += 32) {
    __builtin_prefetch(btRow + n0 + 64, 0, 0);
    v16h bb = loadB16(btRow, half, n0);
    int kA = n0 + 8 * half, kB = n0 + 16 + 8 * half;
    v16h a0, a1;
#pragma unroll
    for (int e = 0; e < 8; ++e) {
      a0[e]     = (f16)(ar0[kA + e] * inv0);
      a0[8 + e] = (f16)(ar0[kB + e] * inv0);
      a1[e]     = (f16)(ar1[kA + e] * inv1);
      a1[8 + e] = (f16)(ar1[kB + e] * inv1);
    }
    c0 = wmma16(a0, bb, c0);
    c1 = wmma16(a1, bb, c1);
  }
#pragma unroll
  for (int v = 0; v < 8; ++v) {
    int mr = v + 8 * half;
    upd[((size_t)b * 32 + mr) * HH + hh] = c0[v];
    upd[((size_t)b * 32 + mr + 16) * HH + hh] = c1[v];
  }
}

// ---------------------------------------------------------------------------
// level-sum -> GRU -> LN -> MLP -> residual -> unit-normalize -> new slots
// grid = BB, block = 128 (tiny: 8 objects x 128)
// ---------------------------------------------------------------------------
__global__ void k_gru(const float* __restrict__ upd, const float* __restrict__ gh_pre,
                      const float* __restrict__ gru_wi, const float* __restrict__ gru_bi,
                      const float* __restrict__ obj_dirs,
                      const float* __restrict__ norm_g, const float* __restrict__ norm_b,
                      const float* __restrict__ mlp_w1, const float* __restrict__ mlp_b1,
                      const float* __restrict__ mlp_w2, const float* __restrict__ mlp_b2,
                      float* __restrict__ slots, float* __restrict__ slots_out) {
  __shared__ float ou[NOBJ][HH];
  __shared__ float nd[NOBJ][HH];
  __shared__ float tt[NOBJ][HH];
  __shared__ float m1[NOBJ][HH];
  __shared__ float onorm[NOBJ], omu[NOBJ], osg[NOBJ];
  int tid = threadIdx.x;
  int b = blockIdx.x;
  const float* ub = upd + (size_t)b * 32 * HH;
  for (int idx = tid; idx < NOBJ * HH; idx += blockDim.x) {
    int o = idx >> 7, h = idx & 127;
    ou[o][h] = ub[(o * 3 + 0) * HH + h] + ub[(o * 3 + 1) * HH + h] +
               ub[(o * 3 + 2) * HH + h];
  }
  __syncthreads();
  for (int idx = tid; idx < NOBJ * HH; idx += blockDim.x) {
    int o = idx >> 7, h = idx & 127;
    float ir = gru_bi[h], iz = gru_bi[h + HH], in = gru_bi[h + 2 * HH];
    const float* wr = gru_wi + (size_t)h * HH;
    const float* wz = gru_wi + (size_t)(h + HH) * HH;
    const float* wn = gru_wi + (size_t)(h + 2 * HH) * HH;
    for (int k = 0; k < HH; ++k) {
      float u = ou[o][k];
      ir += u * wr[k];
      iz += u * wz[k];
      in += u * wn[k];
    }
    float hr = gh_pre[o * 3 * HH + h];
    float hz = gh_pre[o * 3 * HH + h + HH];
    float hn = gh_pre[o * 3 * HH + h + 2 * HH];
    float r = 1.0f / (1.0f + expf(-(ir + hr)));
    float z = 1.0f / (1.0f + expf(-(iz + hz)));
    float nnv = tanhf(in + r * hn);
    nd[o][h] = (1.0f - z) * nnv + z * obj_dirs[o * HH + h];
  }
  __syncthreads();
  if (tid < NOBJ) {
    float s = 0, s2 = 0;
    for (int h = 0; h < HH; ++h) { float v = nd[tid][h]; s += v; s2 += v * v; }
    float mu = s * (1.0f / HH);
    omu[tid] = mu;
    osg[tid] = rsqrtf(fmaxf(s2 * (1.0f / HH) - mu * mu, 0.f) + 1e-5f);
  }
  __syncthreads();
  for (int idx = tid; idx < NOBJ * HH; idx += blockDim.x) {
    int o = idx >> 7, h = idx & 127;
    tt[o][h] = (nd[o][h] - omu[o]) * osg[o] * norm_g[h] + norm_b[h];
  }
  __syncthreads();
  for (int idx = tid; idx < NOBJ * HH; idx += blockDim.x) {
    int o = idx >> 7, h = idx & 127;
    float acc = mlp_b1[h];
    for (int k = 0; k < HH; ++k) acc += tt[o][k] * mlp_w1[(size_t)k * HH + h];
    m1[o][h] = 0.5f * acc * (1.0f + erff(acc * 0.70710678118f));
  }
  __syncthreads();
  for (int idx = tid; idx < NOBJ * HH; idx += blockDim.x) {
    int o = idx >> 7, h = idx & 127;
    float acc = mlp_b2[h];
    for (int k = 0; k < HH; ++k) acc += m1[o][k] * mlp_w2[(size_t)k * HH + h];
    nd[o][h] += 0.2f * acc;
  }
  __syncthreads();
  if (tid < NOBJ) {
    float s2 = 0;
    for (int h = 0; h < HH; ++h) { float v = nd[tid][h]; s2 += v * v; }
    onorm[tid] = 1.0f / fmaxf(sqrtf(s2), 1e-12f);
  }
  __syncthreads();
  const float radii[3] = {0.2f, 0.5f, 0.8f};
  for (int idx = tid; idx < NSLOT * HH; idx += blockDim.x) {
    int k = idx >> 7, h = idx & 127;
    int o = k / 3, l = k % 3;
    float v = nd[o][h] * onorm[o] * radii[l];
    slots[((size_t)b * NSLOT + k) * HH + h] = v;
    slots_out[((size_t)b * NSLOT + k) * HH + h] = v;
  }
}

// ---------------------------------------------------------------------------
// host launch
// ---------------------------------------------------------------------------
extern "C" void kernel_launch(void* const* d_in, const int* in_sizes, int n_in,
                              void* d_out, int out_size, void* d_ws, size_t ws_size,
                              hipStream_t stream) {
  const float* x            = (const float*)d_in[0];
  const float* fe_w1        = (const float*)d_in[1];
  const float* fe_b1        = (const float*)d_in[2];
  const float* ln1_g        = (const float*)d_in[3];
  const float* ln1_b        = (const float*)d_in[4];
  const float* fe_w2        = (const float*)d_in[5];
  const float* fe_b2        = (const float*)d_in[6];
  const float* ln2_g        = (const float*)d_in[7];
  const float* ln2_b        = (const float*)d_in[8];
  const float* hp_w         = (const float*)d_in[9];
  const float* hp_b         = (const float*)d_in[10];
  const float* obj_dirs     = (const float*)d_in[11];
  const float* radius_scale = (const float*)d_in[12];
  const float* gru_wi       = (const float*)d_in[13];
  const float* gru_wh       = (const float*)d_in[14];
  const float* gru_bi       = (const float*)d_in[15];
  const float* gru_bh       = (const float*)d_in[16];
  const float* mlp_w1       = (const float*)d_in[17];
  const float* mlp_b1       = (const float*)d_in[18];
  const float* mlp_w2       = (const float*)d_in[19];
  const float* mlp_b2       = (const float*)d_in[20];
  const float* norm_g       = (const float*)d_in[21];
  const float* norm_b       = (const float*)d_in[22];
  (void)in_sizes; (void)n_in; (void)out_size; (void)ws_size;

  float* out_slots = (float*)d_out;
  float* out_attn  = out_slots + (size_t)BB * NSLOT * HH;
  float* out_dens  = out_attn + (size_t)BB * NSLOT * NN;

  char* ws = (char*)d_ws;
  size_t off = 0;
  auto take = [&](size_t bytes) -> char* {
    off = (off + 255) & ~(size_t)255;
    char* p = ws + off;
    off += bytes;
    return p;
  };
  f16*   xh      = (f16*)take((size_t)BB * NN * DD * 2);
  float* x2      = (float*)take((size_t)BB * NN * 4);
  f16*   w1t     = (f16*)take((size_t)HH * DD * 2);
  f16*   w2t     = (f16*)take((size_t)HH * HH * 2);
  f16*   hpt     = (f16*)take((size_t)HH * HH * 2);
  f16*   feats_h = (f16*)take((size_t)BB * NN * HH * 2);
  f16*   featsT  = (f16*)take((size_t)BB * HH * NN * 2);
  float* f2b     = (float*)take((size_t)BB * NN * 4);
  float* frb     = (float*)take((size_t)BB * NN * 4);
  float* slots   = (float*)take((size_t)BB * NSLOT * HH * 4);
  float* updb    = (float*)take((size_t)BB * 32 * HH * 4);
  float* asum    = (float*)take((size_t)BB * NSLOT * 4);
  float* ghp     = (float*)take((size_t)NOBJ * 3 * HH * 4);
  float* dotsb   = (float*)take((size_t)BB * NSLOT * NN * 4);
  float* snorm   = (float*)take((size_t)BB * NSLOT * 2 * 4);

  k_prep_x<<<(BB * NN + 255) / 256, 256, 0, stream>>>(x, xh, x2);
  k_transpose_f16<<<(DD * HH + 255) / 256, 256, 0, stream>>>(fe_w1, w1t, DD, HH);
  k_transpose_f16<<<(HH * HH + 255) / 256, 256, 0, stream>>>(fe_w2, w2t, HH, HH);
  k_transpose_f16<<<(HH * HH + 255) / 256, 256, 0, stream>>>(hp_w, hpt, HH, HH);
  k_init<<<1, 256, 0, stream>>>(obj_dirs, gru_wh, gru_bh, slots, ghp);
  k_density<<<BB * (NN / 16), 32, 0, stream>>>(xh, x2, out_dens);
  k_encoder<<<BB * (NN / 16), 256, 0, stream>>>(xh, fe_b1, ln1_g, ln1_b, fe_b2, ln2_g,
                                                ln2_b, hp_b, w1t, w2t, hpt, out_dens,
                                                feats_h, featsT, f2b, frb);
  for (int it = 0; it < 3; ++it) {
    k_zero<<<2, 256, 0, stream>>>(asum, BB * NSLOT);
    k_dots<<<BB * (NN / 128), 256, 0, stream>>>(slots, feats_h, dotsb, snorm);
    k_attn<<<BB * (NN / 256), 256, 0, stream>>>(dotsb, snorm, f2b, frb, out_dens,
                                                radius_scale, out_attn, asum);
    k_upd<<<BB, 256, 0, stream>>>(out_attn, asum, featsT, updb);
    k_gru<<<BB, 128, 0, stream>>>(updb, ghp, gru_wi, gru_bi, obj_dirs, norm_g, norm_b,
                                  mlp_w1, mlp_b1, mlp_w2, mlp_b2, slots, out_slots);
  }
}
